// PairContrastiveLoss_69174743269977
// MI455X (gfx1250) — matrix operand order repaired and checked by reading, not compiled, not run on hardware
//
#include <hip/hip_runtime.h>
#include <hip/hip_bf16.h>

// ---------------------------------------------------------------------------
// Problem constants (match reference)
// ---------------------------------------------------------------------------
#define BB        2
#define C_IN      32
#define HH        512
#define WW        512
#define HIDDEN    768
#define SIZE      16
#define LATENT    128
#define NUM_CLASS 21
#define HP        32
#define WP        32
#define NPATCH    (HP * WP)              // 1024
#define KIM       (C_IN * SIZE * SIZE)   // 8192
#define LC        (LATENT * NUM_CLASS)   // 2688
#define SPLITK    4
#define KSPL      (KIM / SPLITK)         // 2048

typedef __attribute__((ext_vector_type(16))) __bf16 v16bf;
typedef __attribute__((ext_vector_type(8)))  float  v8f;

union V16 {
    v16bf v;
    uint4 u[2];
};

static __device__ __forceinline__ unsigned short f32_to_bf16_bits(float f) {
    unsigned int u = __float_as_uint(f);
    u += 0x7FFFu + ((u >> 16) & 1u);      // round-to-nearest-even
    return (unsigned short)(u >> 16);
}

// ---------------------------------------------------------------------------
// (16*MT) x (16*NT) output tile per wave, bf16 16x16x32 WMMA, f32 accumulate.
// A: row-major over K (M rows).  B: row-major over K (N rows)  (NT GEMM).
// LDK is compile-time so every fragment is addressed as ONE base pointer +
// constant immediate offset (i*16*LDK*2 bytes < 2^23) -> global_load_b128
// with 24-bit ioffset, only two live pointers, no spills.
//   A frag lane L: row = 16*i + (L&15), two 8-bf16 runs at k+8*(L>>4), +16
//   B frag lane L: col = 16*j + (L&15), two 8-bf16 runs at k+8*(L>>4), +16
// ---------------------------------------------------------------------------
template <int MT, int NT, int LDK>
static __device__ __forceinline__ void wmma_tile(
    const unsigned short* __restrict__ Abase,   // &A[row0 * LDK + kbegin]
    const unsigned short* __restrict__ Bbase,   // &B[col0 * LDK + kbegin]
    int K, int lane, v8f acc[MT][NT])
{
    const int r  = lane & 15;
    const int hi = lane >> 4;

    const unsigned short* a = Abase + (size_t)r * LDK + hi * 8;
    const unsigned short* b = Bbase + (size_t)r * LDK + hi * 8;

    for (int k0 = 0; k0 < K; k0 += 32) {
        V16 Af[MT], Bf[NT];
        #pragma unroll
        for (int i = 0; i < MT; ++i) {
            Af[i].u[0] = *(const uint4*)(a + i * 16 * LDK);        // k0 + 8*hi
            Af[i].u[1] = *(const uint4*)(a + i * 16 * LDK + 16);   // k0 + 16 + 8*hi
        }
        #pragma unroll
        for (int j = 0; j < NT; ++j) {
            Bf[j].u[0] = *(const uint4*)(b + j * 16 * LDK);        // k0 + 8*hi
            Bf[j].u[1] = *(const uint4*)(b + j * 16 * LDK + 16);
        }
        #pragma unroll
        for (int i = 0; i < MT; ++i)
            #pragma unroll
            for (int j = 0; j < NT; ++j)
                acc[i][j] = __builtin_amdgcn_wmma_f32_16x16x32_bf16(
                    false, Af[i].v, false, Bf[j].v, (short)0, acc[i][j], false, false);
        a += 32;
        b += 32;
    }
}

// ---------------------------------------------------------------------------
// 1) im2col of x -> bf16 matrix Xc[b][n][k], k = c*256 + i*16 + j
// ---------------------------------------------------------------------------
__global__ void im2col_bf16_kernel(const float* __restrict__ x,
                                   unsigned short* __restrict__ Xc)
{
    size_t idx = (size_t)blockIdx.x * 256 + threadIdx.x;   // B*N*K = 16777216
    int k = (int)(idx & (KIM - 1));
    int n = (int)((idx >> 13) & (NPATCH - 1));
    int b = (int)(idx >> 23);
    int c  = k >> 8;
    int ij = k & 255;
    int i  = ij >> 4;
    int j  = ij & 15;
    int ph = n >> 5;
    int pw = n & 31;
    float v = x[(((size_t)b * C_IN + c) * HH + (ph * SIZE + i)) * WW + (pw * SIZE + j)];
    Xc[idx] = f32_to_bf16_bits(v);
}

// ---------------------------------------------------------------------------
// 2) generic f32 -> bf16 convert (conv_w already [d][k]-major; latent [l][d])
// ---------------------------------------------------------------------------
__global__ void cvt_bf16_kernel(const float* __restrict__ src,
                                unsigned short* __restrict__ dst, int n)
{
    int i = blockIdx.x * 256 + threadIdx.x;
    if (i < n) dst[i] = f32_to_bf16_bits(src[i]);
}

// ---------------------------------------------------------------------------
// 3) row L2 norms (one block / row)
// ---------------------------------------------------------------------------
__global__ void row_norm_kernel(const float* __restrict__ src,
                                float* __restrict__ dst, int ncols)
{
    __shared__ float sh[256];
    const float* row = src + (size_t)blockIdx.x * ncols;
    float s = 0.0f;
    for (int i = threadIdx.x; i < ncols; i += 256) {
        float v = row[i];
        s += v * v;
    }
    sh[threadIdx.x] = s;
    __syncthreads();
    for (int off = 128; off > 0; off >>= 1) {
        if ((int)threadIdx.x < off) sh[threadIdx.x] += sh[threadIdx.x + off];
        __syncthreads();
    }
    if (threadIdx.x == 0) dst[blockIdx.x] = sqrtf(sh[0]);
}

// ---------------------------------------------------------------------------
// 4) GEMM1 (split-K=4): partial[s][b][n][d] = Xc[b][n][ks..] . conv_w[d][ks..]
//    64x64 tile / wave, 1536 wave-tiles (2 b * 16 m * 12 n * 4 splits)
// ---------------------------------------------------------------------------
__global__ void __launch_bounds__(128, 1)
gemm1_patch_kernel(const unsigned short* __restrict__ Xc,
                   const unsigned short* __restrict__ Wt,
                   float* __restrict__ partial)
{
    const int wid  = threadIdx.x >> 5;
    const int lane = threadIdx.x & 31;
    const int tile = blockIdx.x * 4 + wid;     // 0..1535
    const int s  = tile & 3;
    const int t  = tile >> 2;                  // 0..383
    const int nt = t % 12;
    const int t2 = t / 12;
    const int mt = t2 % 16;
    const int b  = t2 / 16;
    const int n0 = mt * 64;
    const int d0 = nt * 64;

    const unsigned short* A  = Xc + ((size_t)b * NPATCH + n0) * KIM + s * KSPL;
    const unsigned short* Bm = Wt + (size_t)d0 * KIM + s * KSPL;

    v8f acc[4][4];
    #pragma unroll
    for (int i = 0; i < 4; ++i)
        #pragma unroll
        for (int j = 0; j < 4; ++j) acc[i][j] = (v8f)0.0f;

    wmma_tile<4, 4, KIM>(A, Bm, KSPL, lane, acc);

    const int r  = lane & 15;
    const int hi = lane >> 4;
    float* dst = partial + (size_t)s * BB * NPATCH * HIDDEN;
    #pragma unroll
    for (int ti = 0; ti < 4; ++ti)
        #pragma unroll
        for (int tj = 0; tj < 4; ++tj)
            #pragma unroll
            for (int e = 0; e < 8; ++e) {
                int m = n0 + ti * 16 + e + 8 * hi;   // patch index
                int d = d0 + tj * 16 + r;            // hidden dim
                dst[((size_t)b * NPATCH + m) * HIDDEN + d] = acc[ti][tj][e];
            }
}

// ---------------------------------------------------------------------------
// 5) split-K reduce + bias -> patch f32 (for norms) and bf16 (for GEMM2)
// ---------------------------------------------------------------------------
__global__ void splitk_reduce_kernel(const float* __restrict__ partial,
                                     const float* __restrict__ convb,
                                     float* __restrict__ patchF,
                                     unsigned short* __restrict__ patchB)
{
    int idx = blockIdx.x * 256 + threadIdx.x;   // 2*1024*768 = 1572864
    int d = idx % HIDDEN;
    const size_t S = (size_t)BB * NPATCH * HIDDEN;
    float v = convb[d];
    v += partial[idx];
    v += partial[idx + S];
    v += partial[idx + 2 * S];
    v += partial[idx + 3 * S];
    patchF[idx] = v;
    patchB[idx] = f32_to_bf16_bits(v);
}

// ---------------------------------------------------------------------------
// 6) GEMM2: sim[b][l][n] = (lat[l][:].patch[b][n][:]) / max(|lat|*|pat|, eps)
//    64x64 tile / wave, 1344 wave-tiles (2 b * 42 l * 16 n)
// ---------------------------------------------------------------------------
__global__ void __launch_bounds__(128, 1)
gemm2_sim_kernel(const unsigned short* __restrict__ latB,
                 const unsigned short* __restrict__ patchB,
                 const float* __restrict__ latN,
                 const float* __restrict__ patN,
                 float* __restrict__ simF)
{
    const int wid  = threadIdx.x >> 5;
    const int lane = threadIdx.x & 31;
    const int tile = blockIdx.x * 4 + wid;      // 0..1343
    const int nt = tile % 16;
    const int t2 = tile / 16;
    const int lt = t2 % 42;
    const int b  = t2 / 42;
    const int l0 = lt * 64;
    const int n0 = nt * 64;

    const unsigned short* A  = latB + (size_t)l0 * HIDDEN;
    const unsigned short* Bm = patchB + ((size_t)b * NPATCH + n0) * HIDDEN;

    v8f acc[4][4];
    #pragma unroll
    for (int i = 0; i < 4; ++i)
        #pragma unroll
        for (int j = 0; j < 4; ++j) acc[i][j] = (v8f)0.0f;

    wmma_tile<4, 4, HIDDEN>(A, Bm, HIDDEN, lane, acc);

    const int r  = lane & 15;
    const int hi = lane >> 4;
    #pragma unroll
    for (int ti = 0; ti < 4; ++ti)
        #pragma unroll
        for (int tj = 0; tj < 4; ++tj)
            #pragma unroll
            for (int e = 0; e < 8; ++e) {
                int l = l0 + ti * 16 + e + 8 * hi;
                int n = n0 + tj * 16 + r;
                float denom = fmaxf(latN[l] * patN[b * NPATCH + n], 1e-8f);
                simF[((size_t)b * LC + l) * NPATCH + n] = acc[ti][tj][e] / denom;
            }
}

// ---------------------------------------------------------------------------
// 7) per-class mean over the 128 latents
// ---------------------------------------------------------------------------
__global__ void class_mean_kernel(const float* __restrict__ simF,
                                  float* __restrict__ classSim)
{
    int idx = blockIdx.x * 256 + threadIdx.x;       // B*21*1024 = 43008
    if (idx >= BB * NUM_CLASS * NPATCH) return;
    int n = idx % NPATCH;
    int c = (idx / NPATCH) % NUM_CLASS;
    int b = idx / (NUM_CLASS * NPATCH);
    const float* base = simF + ((size_t)b * LC + c) * NPATCH + n;
    float s = 0.0f;
    #pragma unroll 4
    for (int j = 0; j < LATENT; ++j)
        s += base[(size_t)j * NUM_CLASS * NPATCH];
    classSim[idx] = s * (1.0f / (float)LATENT);
}

// ---------------------------------------------------------------------------
// 8) masked sum of exp(sim / TEMP), out = -log(s)   (TEMP = 0.5 -> *2)
// ---------------------------------------------------------------------------
__global__ void final_loss_kernel(const float* __restrict__ classSim,
                                  const int* __restrict__ mask,
                                  float* __restrict__ out)
{
    __shared__ float sh[256];
    float s = 0.0f;
    for (int i = threadIdx.x; i < BB * NUM_CLASS * NPATCH; i += 256)
        if (mask[i] != 0) s += expf(classSim[i] * 2.0f);
    sh[threadIdx.x] = s;
    __syncthreads();
    for (int off = 128; off > 0; off >>= 1) {
        if ((int)threadIdx.x < off) sh[threadIdx.x] += sh[threadIdx.x + off];
        __syncthreads();
    }
    if (threadIdx.x == 0) out[0] = -logf(sh[0]);
}

// ---------------------------------------------------------------------------
// Launch
// ---------------------------------------------------------------------------
extern "C" void kernel_launch(void* const* d_in, const int* in_sizes, int n_in,
                              void* d_out, int out_size, void* d_ws, size_t ws_size,
                              hipStream_t stream)
{
    (void)in_sizes; (void)n_in; (void)out_size; (void)ws_size;

    const float* x      = (const float*)d_in[0];
    const int*   mask   = (const int*)  d_in[1];
    const float* conv_w = (const float*)d_in[2];
    const float* conv_b = (const float*)d_in[3];
    const float* latent = (const float*)d_in[4];
    float*       out    = (float*)d_out;

    // Workspace carve-up (all sections 256B aligned).
    // split-K partials and simF alias the same region (disjoint live ranges).
    char* ws = (char*)d_ws;
    size_t off = 0;
    unsigned short* XcBF   = (unsigned short*)(ws + off); off += (size_t)BB * NPATCH * KIM * 2;      // 32 MB
    unsigned short* WtBF   = (unsigned short*)(ws + off); off += (size_t)HIDDEN * KIM * 2;           // 12.6 MB
    unsigned short* latBF  = (unsigned short*)(ws + off); off += (size_t)LC * HIDDEN * 2;            // 4.1 MB
    float*          patchF = (float*)(ws + off);          off += (size_t)BB * NPATCH * HIDDEN * 4;   // 6.3 MB
    unsigned short* patchB = (unsigned short*)(ws + off); off += (size_t)BB * NPATCH * HIDDEN * 2;   // 3.1 MB
    float*          latN   = (float*)(ws + off);          off += (size_t)LC * 4;
    float*          patN   = (float*)(ws + off);          off += (size_t)BB * NPATCH * 4;
    float*          shared = (float*)(ws + off);          off += (size_t)SPLITK * BB * NPATCH * HIDDEN * 4; // 25.2 MB
    float*          clsSim = (float*)(ws + off);          off += (size_t)BB * NUM_CLASS * NPATCH * 4;

    float* partial = shared;   // live: steps 4-5
    float* simF    = shared;   // live: steps 6-7

    // 1) im2col (bf16)
    im2col_bf16_kernel<<<(BB * NPATCH * KIM) / 256, 256, 0, stream>>>(x, XcBF);

    // 2) bf16 conversions
    cvt_bf16_kernel<<<(HIDDEN * KIM) / 256, 256, 0, stream>>>(conv_w, WtBF, HIDDEN * KIM);
    cvt_bf16_kernel<<<(LC * HIDDEN + 255) / 256, 256, 0, stream>>>(latent, latBF, LC * HIDDEN);

    // 3) latent norms (f32 inputs)
    row_norm_kernel<<<LC, 256, 0, stream>>>(latent, latN, HIDDEN);

    // 4) patch-embedding GEMM, WMMA bf16, 64x64 tiles, split-K=4
    gemm1_patch_kernel<<<(BB * 16 * 12 * SPLITK) / 4, 128, 0, stream>>>(XcBF, WtBF, partial);

    // 5) deterministic split-K reduce + bias -> patch f32 + bf16
    splitk_reduce_kernel<<<(BB * NPATCH * HIDDEN) / 256, 256, 0, stream>>>(partial, conv_b, patchF, patchB);

    // 6) patch norms (from f32 patch)
    row_norm_kernel<<<BB * NPATCH, 256, 0, stream>>>(patchF, patN, HIDDEN);

    // 7) cosine-similarity GEMM, WMMA bf16, 64x64 tiles
    gemm2_sim_kernel<<<(BB * 42 * 16) / 4, 128, 0, stream>>>(latBF, patchB, latN, patN, simF);

    // 8) mean over latents per class
    class_mean_kernel<<<(BB * NUM_CLASS * NPATCH) / 256, 256, 0, stream>>>(simF, clsSim);

    // 9) masked exp-sum -> -log
    final_loss_kernel<<<1, 256, 0, stream>>>(clsSim, mask, out);
}
